// MultiheadAttention_76879914598863
// MI455X (gfx1250) — matrix operand order repaired and checked by reading, not compiled, not run on hardware
//
#include <hip/hip_runtime.h>

#define EMBED   1024
#define HEADS   16
#define HDIM    64
#define BATCH   2
#define SEQ     2048
#define NTOK    (BATCH * SEQ)   // 4096

typedef __attribute__((ext_vector_type(16))) __bf16        v16bf;
typedef __attribute__((ext_vector_type(8)))  float         v8f;
typedef __attribute__((ext_vector_type(4)))  unsigned int  u32x4;
typedef __attribute__((ext_vector_type(8)))  int           i32x8;
typedef __attribute__((ext_vector_type(4)))  int           i32x4;

union B16x16 { uint4 q[2]; v16bf v; };

#if defined(__has_builtin)
#  if __has_builtin(__builtin_amdgcn_tensor_load_to_lds)
#    define HAVE_TDM 1
#  else
#    define HAVE_TDM 0
#  endif
#else
#  define HAVE_TDM 0
#endif

__device__ __forceinline__ unsigned short f2bf(float f) {
    unsigned int u = __float_as_uint(f);
    u += 0x7FFFu + ((u >> 16) & 1u);           // round-to-nearest-even
    return (unsigned short)(u >> 16);
}

#if HAVE_TDM
__device__ __forceinline__ void tdm_issue(u32x4 g0, i32x8 g1) {
    i32x4 z4 = (i32x4){0, 0, 0, 0};
#if defined(__clang_major__) && (__clang_major__ >= 23)
    i32x8 z8 = (i32x8){0, 0, 0, 0, 0, 0, 0, 0};
    __builtin_amdgcn_tensor_load_to_lds(g0, g1, z4, z4, z8, 0);
#else
    __builtin_amdgcn_tensor_load_to_lds(g0, g1, z4, z4, 0);
#endif
}

// 2D bf16 tile DMA: tensor is [tensor_d1][tensor_d0] (d0 contiguous, stride0
// elements between rows); tile is [tile_d1][tile_d0] starting at gaddr; rows
// packed contiguously into LDS at lds_off (stride tile_d0).  D# per ISA §8.
__device__ __forceinline__ void tdm_load_2d(const void* gaddr, unsigned lds_off,
                                            unsigned tensor_d0, unsigned tensor_d1,
                                            unsigned stride0,
                                            unsigned tile_d0, unsigned tile_d1) {
    unsigned long long ga = (unsigned long long)(size_t)gaddr;
    u32x4 g0;
    g0[0] = 0x1u;                                   // count=1, user mode
    g0[1] = lds_off;                                // lds_addr (bytes)
    g0[2] = (unsigned)ga;                           // global_addr[31:0]
    g0[3] = (unsigned)(ga >> 32) | (2u << 30);      // addr[56:32] | type=2
    i32x8 g1;
    g1[0] = (int)(1u << 16);                        // data_size=1 (2 bytes)
    g1[1] = (int)((tensor_d0 & 0xFFFFu) << 16);     // tensor_dim0[15:0]
    g1[2] = (int)((tensor_d0 >> 16) | ((tensor_d1 & 0xFFFFu) << 16));
    g1[3] = (int)((tensor_d1 >> 16) | (tile_d0 << 16));
    g1[4] = (int)tile_d1;                           // tile_dim1 | tile_dim2=0
    g1[5] = (int)stride0;                           // tensor_dim0_stride[31:0]
    g1[6] = 0;                                      // stride0[47:32]|stride1 lo
    g1[7] = 0;
    tdm_issue(g0, g1);
}
#endif // HAVE_TDM

// A-matrix 16x32 bf16 fragment (ISA 7.12.2): lane(g=lane/16, m=lane%16) holds
// K = k0+8g..+7 (halves 0..7) and K = k0+16+8g..+7 (halves 8..15), row m.
__device__ __forceinline__ v16bf load_afrag(const unsigned short* p, int stride,
                                            int row, int k0, int g) {
    B16x16 u;
    const unsigned short* base = p + row * stride + k0;
    u.q[0] = *(const uint4*)(base + 8 * g);
    u.q[1] = *(const uint4*)(base + 16 + 8 * g);
    return u.v;
}

// B-matrix 32x16 bf16 fragment: lane(g, n=lane%16) holds column n,
// K = k0+8g..+7 (halves 0..7) and K = k0+16+8g..+7 (halves 8..15).
__device__ __forceinline__ v16bf load_bfrag(const unsigned short* p, int stride,
                                            int row, int k0, int g) {
    B16x16 u;
    const unsigned short* base = p + row * stride + k0;
    u.q[0] = *(const uint4*)(base + 8 * g);
    u.q[1] = *(const uint4*)(base + 16 + 8 * g);
    return u.v;
}

__device__ __forceinline__ v8f wmma_bf16(v16bf a, v16bf b, v8f c) {
    return __builtin_amdgcn_wmma_f32_16x16x32_bf16(false, a, false, b,
                                                   (short)0, c, false, false);
}

// ---------------------------------------------------------------------------
// Kernel 1: fused QKV projection.  Y = X @ W^T + b, downconvert to bf16.
// Q,K stored [B,H,S,Dh]; V stored transposed [B,H,Dh,S] so P*V is NT later.
// Grid: (NTOK/64, EMBED/128, 3).  Block: 256 threads = 8 waves (2m x 4n).
// ---------------------------------------------------------------------------
__global__ __launch_bounds__(256) void qkv_proj_kernel(
    const float* __restrict__ X,
    const float* __restrict__ Wq, const float* __restrict__ bq,
    const float* __restrict__ Wk, const float* __restrict__ bk,
    const float* __restrict__ Wv, const float* __restrict__ bv,
    unsigned short* __restrict__ Qb, unsigned short* __restrict__ Kb,
    unsigned short* __restrict__ Vt)
{
    __shared__ __align__(16) unsigned short sA[64 * 64];    // 8 KB
    __shared__ __align__(16) unsigned short sB[128 * 64];   // 16 KB

    const int mt = blockIdx.x, nt = blockIdx.y, which = blockIdx.z;
    const float* W    = (which == 0) ? Wq : (which == 1) ? Wk : Wv;
    const float* bias = (which == 0) ? bq : (which == 1) ? bk : bv;

    const int tid  = threadIdx.x;
    const int lane = tid & 31, wave = tid >> 5;
    const int wm = wave & 1, wn = wave >> 1;
    const int g = lane >> 4, hl = lane & 15;

    v8f acc[2][2];
#pragma unroll
    for (int i = 0; i < 2; ++i)
#pragma unroll
        for (int j = 0; j < 2; ++j) acc[i][j] = (v8f){0,0,0,0,0,0,0,0};

    for (int kc = 0; kc < EMBED; kc += 64) {
        // Stage A: X[mt*64 .. +63][kc .. +63] fp32 -> bf16 LDS
#pragma unroll
        for (int i = 0; i < 4; ++i) {
            int idx = tid + i * 256;            // 1024 float4 total
            int r = idx >> 4, c4 = idx & 15;
            float4 f = *(const float4*)&X[(size_t)(mt * 64 + r) * EMBED + kc + c4 * 4];
            int o = r * 64 + c4 * 4;
            sA[o + 0] = f2bf(f.x); sA[o + 1] = f2bf(f.y);
            sA[o + 2] = f2bf(f.z); sA[o + 3] = f2bf(f.w);
        }
        // Stage B: W[nt*128 .. +127][kc .. +63] fp32 -> bf16 LDS
#pragma unroll
        for (int i = 0; i < 8; ++i) {
            int idx = tid + i * 256;            // 2048 float4 total
            int r = idx >> 4, c4 = idx & 15;
            float4 f = *(const float4*)&W[(size_t)(nt * 128 + r) * EMBED + kc + c4 * 4];
            int o = r * 64 + c4 * 4;
            sB[o + 0] = f2bf(f.x); sB[o + 1] = f2bf(f.y);
            sB[o + 2] = f2bf(f.z); sB[o + 3] = f2bf(f.w);
        }
        __syncthreads();

#pragma unroll
        for (int ks = 0; ks < 64; ks += 32) {
            v16bf af[2], bf[2];
#pragma unroll
            for (int i = 0; i < 2; ++i)
                af[i] = load_afrag(sA, 64, wm * 32 + i * 16 + hl, ks, g);
#pragma unroll
            for (int j = 0; j < 2; ++j)
                bf[j] = load_bfrag(sB, 64, wn * 32 + j * 16 + hl, ks, g);
#pragma unroll
            for (int i = 0; i < 2; ++i)
#pragma unroll
                for (int j = 0; j < 2; ++j)
                    acc[i][j] = wmma_bf16(af[i], bf[j], acc[i][j]);
        }
        __syncthreads();
    }

    // Epilogue: C-layout lane(g,c): VGPR r -> (M = r + 8g, N = c)
    const int c = lane & 15;
#pragma unroll
    for (int i = 0; i < 2; ++i) {
#pragma unroll
        for (int j = 0; j < 2; ++j) {
            int n = nt * 128 + wn * 32 + j * 16 + c;
            float bvn = bias[n];
            int h = n >> 6, d = n & 63;
#pragma unroll
            for (int r = 0; r < 8; ++r) {
                int row = mt * 64 + wm * 32 + i * 16 + r + 8 * g;  // token idx
                int b = row >> 11, s = row & (SEQ - 1);
                unsigned short v = f2bf(acc[i][j][r] + bvn);
                if (which == 2)
                    Vt[((size_t)(b * HEADS + h) * HDIM + d) * SEQ + s] = v;
                else if (which == 1)
                    Kb[((size_t)(b * HEADS + h) * SEQ + s) * HDIM + d] = v;
                else
                    Qb[((size_t)(b * HEADS + h) * SEQ + s) * HDIM + d] = v;
            }
        }
    }
}

// ---------------------------------------------------------------------------
// Kernel 2: flash attention.  Grid (SEQ/64, B*HEADS), block 128 = 4 waves.
// K / V^T chunks double-buffered in LDS via the Tensor Data Mover: wave 0
// prefetches chunk i+1 while all waves compute on chunk i; in-order TENSORcnt
// (wait <= 2) guarantees the current chunk's two DMAs have landed.
// ---------------------------------------------------------------------------
__global__ __launch_bounds__(128) void attn_kernel(
    const unsigned short* __restrict__ Qb, const unsigned short* __restrict__ Kb,
    const unsigned short* __restrict__ Vt, const float* __restrict__ mask,
    unsigned short* __restrict__ ctx)
{
    __shared__ __align__(16) unsigned short sK[2][64 * 64];   // 16 KB
    __shared__ __align__(16) unsigned short sV[2][64 * 64];   // 16 KB (V^T rows = d)
    __shared__ __align__(16) unsigned short sP[4][16 * 64];   // 8 KB  per-wave P

    const int bh = blockIdx.y;
    const int b = bh >> 4, h = bh & 15;
    const int qt = blockIdx.x;
    const int tid = threadIdx.x, lane = tid & 31, wave = tid >> 5;
    const int g = lane >> 4, hl = lane & 15;

    const unsigned short* Qbase = Qb + (size_t)bh * SEQ * HDIM;
    const unsigned short* Kbase = Kb + (size_t)bh * SEQ * HDIM;
    const unsigned short* Vbase = Vt + (size_t)bh * HDIM * SEQ;
    const float* maskb = mask + (size_t)b * SEQ;

    // Q fragments for this wave's 16 rows (k = 0..63), loaded once from global.
    const int qrow = qt * 64 + wave * 16 + hl;
    v16bf qf[2];
    qf[0] = load_afrag(Qbase, HDIM, qrow, 0,  g);
    qf[1] = load_afrag(Qbase, HDIM, qrow, 32, g);

    v8f O[4];
#pragma unroll
    for (int dt = 0; dt < 4; ++dt) O[dt] = (v8f){0,0,0,0,0,0,0,0};
    float mrow[8], lrow[8];
#pragma unroll
    for (int r = 0; r < 8; ++r) { mrow[r] = -1e30f; lrow[r] = 0.0f; }

#if HAVE_TDM
    if (wave == 0) {   // prologue: chunk 0 -> buffer 0
        tdm_load_2d(Kbase, (unsigned)(size_t)&sK[0][0], HDIM, SEQ, HDIM, HDIM, 64);
        tdm_load_2d(Vbase, (unsigned)(size_t)&sV[0][0], SEQ, HDIM, SEQ, 64, HDIM);
    }
#endif

    for (int kb = 0, it = 0; kb < SEQ; kb += 64, ++it) {
        const int cur = it & 1;
#if HAVE_TDM
        const int nxt = cur ^ 1;
        if (wave == 0) {
            if (kb + 64 < SEQ) {
                // Prefetch chunk i+1 into the alternate buffer (freed by the
                // trailing barrier of iteration i-1), then wait for chunk i:
                // tensor ops complete in order, so tensorcnt<=2 means the
                // current chunk's two DMAs are done while these two fly.
                tdm_load_2d(Kbase + (size_t)(kb + 64) * HDIM,
                            (unsigned)(size_t)&sK[nxt][0],
                            HDIM, SEQ, HDIM, HDIM, 64);
                tdm_load_2d(Vbase + kb + 64,
                            (unsigned)(size_t)&sV[nxt][0],
                            SEQ, HDIM, SEQ, 64, HDIM);
                __builtin_amdgcn_s_wait_tensorcnt((short)2);
            } else {
                __builtin_amdgcn_s_wait_tensorcnt((short)0);
            }
        }
        __syncthreads();
#else
#pragma unroll
        for (int i = 0; i < 4; ++i) {
            int idx = tid + i * 128;            // 512 uint4 total
            int r = idx >> 3, cc = idx & 7;
            *(uint4*)&sK[cur][r * 64 + cc * 8] =
                *(const uint4*)&Kbase[(size_t)(kb + r) * HDIM + cc * 8];
        }
#pragma unroll
        for (int i = 0; i < 4; ++i) {
            int idx = tid + i * 128;
            int r = idx >> 3, cc = idx & 7;
            *(uint4*)&sV[cur][r * 64 + cc * 8] =
                *(const uint4*)&Vbase[(size_t)r * SEQ + kb + cc * 8];
        }
        __syncthreads();
#endif
        const unsigned short* cK = &sK[cur][0];
        const unsigned short* cV = &sV[cur][0];

        // Scores: 4 tiles of 16x16 over 64 keys, K-dim = 64 (two k=32 WMMAs)
        v8f sc[4];
#pragma unroll
        for (int t = 0; t < 4; ++t) {
            v16bf kflo = load_bfrag(cK, 64, 16 * t + hl, 0,  g);
            v16bf kfhi = load_bfrag(cK, 64, 16 * t + hl, 32, g);
            v8f z = (v8f){0,0,0,0,0,0,0,0};
            z = wmma_bf16(qf[0], kflo, z);
            z = wmma_bf16(qf[1], kfhi, z);
            sc[t] = z;
        }

        // Scale 1/sqrt(64) + additive mask (column = key index)
#pragma unroll
        for (int t = 0; t < 4; ++t) {
            float pen = (1.0f - maskb[kb + 16 * t + hl]) * (-100000.0f);
#pragma unroll
            for (int r = 0; r < 8; ++r) sc[t][r] = sc[t][r] * 0.125f + pen;
        }

        // Online softmax: rows live in 16-lane halves; xor 1..8 stays in-half.
        float alpha[8];
#pragma unroll
        for (int r = 0; r < 8; ++r) {
            float v = fmaxf(fmaxf(sc[0][r], sc[1][r]), fmaxf(sc[2][r], sc[3][r]));
#pragma unroll
            for (int m = 1; m < 16; m <<= 1) v = fmaxf(v, __shfl_xor(v, m, 32));
            float nm = fmaxf(mrow[r], v);
            alpha[r] = __expf(mrow[r] - nm);
            mrow[r] = nm;
        }
        float psum[8];
#pragma unroll
        for (int r = 0; r < 8; ++r) psum[r] = 0.0f;
#pragma unroll
        for (int t = 0; t < 4; ++t)
#pragma unroll
            for (int r = 0; r < 8; ++r) {
                float p = __expf(sc[t][r] - mrow[r]);
                sc[t][r] = p;
                psum[r] += p;
            }
#pragma unroll
        for (int r = 0; r < 8; ++r) {
            float v = psum[r];
#pragma unroll
            for (int m = 1; m < 16; m <<= 1) v += __shfl_xor(v, m, 32);
            lrow[r] = lrow[r] * alpha[r] + v;
        }
#pragma unroll
        for (int dt = 0; dt < 4; ++dt)
#pragma unroll
            for (int r = 0; r < 8; ++r) O[dt][r] *= alpha[r];

        // Re-fragment P via per-wave LDS scratch (same-wave DS ops in-order).
#pragma unroll
        for (int t = 0; t < 4; ++t)
#pragma unroll
            for (int r = 0; r < 8; ++r)
                sP[wave][(r + 8 * g) * 64 + 16 * t + hl] = f2bf(sc[t][r]);

        v16bf pf0 = load_afrag(sP[wave], 64, hl, 0,  g);
        v16bf pf1 = load_afrag(sP[wave], 64, hl, 32, g);
#pragma unroll
        for (int dt = 0; dt < 4; ++dt) {
            v16bf vf0 = load_bfrag(cV, 64, 16 * dt + hl, 0,  g);
            v16bf vf1 = load_bfrag(cV, 64, 16 * dt + hl, 32, g);
            O[dt] = wmma_bf16(pf0, vf0, O[dt]);
            O[dt] = wmma_bf16(pf1, vf1, O[dt]);
        }
        __syncthreads();   // frees buffer `cur` for the next prefetch
    }

    // Normalize and store merged-head ctx [B,S,H*Dh] as bf16.
    float inv[8];
#pragma unroll
    for (int r = 0; r < 8; ++r) inv[r] = 1.0f / lrow[r];
#pragma unroll
    for (int dt = 0; dt < 4; ++dt) {
        int d = 16 * dt + hl;
#pragma unroll
        for (int r = 0; r < 8; ++r) {
            int s = qt * 64 + wave * 16 + r + 8 * g;
            ctx[((size_t)(b * SEQ + s)) * EMBED + h * HDIM + d] =
                f2bf(O[dt][r] * inv[r]);
        }
    }
}

// ---------------------------------------------------------------------------
// Kernel 3: output projection + bias + residual (fp32 out to workspace).
// A-tile (ctx, already bf16) double-buffered via TDM: the next k-chunk's DMA
// overlaps the fp32->bf16 Wo staging and the current chunk's WMMAs.
// Grid (NTOK/64, EMBED/128), block 256 = 8 waves (2m x 4n).
// ---------------------------------------------------------------------------
__global__ __launch_bounds__(256) void oproj_kernel(
    const unsigned short* __restrict__ ctx, const float* __restrict__ Wo,
    const float* __restrict__ bo, const float* __restrict__ X,
    float* __restrict__ tmp)
{
    __shared__ __align__(16) unsigned short sA[2][64 * 64];   // 16 KB
    __shared__ __align__(16) unsigned short sB[128 * 64];     // 16 KB

    const int mt = blockIdx.x, nt = blockIdx.y;
    const int tid = threadIdx.x, lane = tid & 31, wave = tid >> 5;
    const int wm = wave & 1, wn = wave >> 1;
    const int g = lane >> 4, hl = lane & 15;

    const unsigned short* ctxRow = ctx + (size_t)(mt * 64) * EMBED;

    v8f acc[2][2];
#pragma unroll
    for (int i = 0; i < 2; ++i)
#pragma unroll
        for (int j = 0; j < 2; ++j) acc[i][j] = (v8f){0,0,0,0,0,0,0,0};

#if HAVE_TDM
    if (wave == 0)     // prologue: k-chunk 0 -> buffer 0
        tdm_load_2d(ctxRow, (unsigned)(size_t)&sA[0][0],
                    EMBED, NTOK, EMBED, 64, 64);
#endif

    for (int kc = 0, it = 0; kc < EMBED; kc += 64, ++it) {
        const int cur = it & 1;
#if HAVE_TDM
        const int nxt = cur ^ 1;
        if (wave == 0 && kc + 64 < EMBED)
            tdm_load_2d(ctxRow + kc + 64, (unsigned)(size_t)&sA[nxt][0],
                        EMBED, NTOK, EMBED, 64, 64);
#else
#pragma unroll
        for (int i = 0; i < 2; ++i) {
            int idx = tid + i * 256;
            int r = idx >> 3, cc = idx & 7;
            *(uint4*)&sA[cur][r * 64 + cc * 8] =
                *(const uint4*)&ctxRow[(size_t)r * EMBED + kc + cc * 8];
        }
#endif
        // B: Wo fp32 -> bf16 (all 8 waves; overlaps both in-flight TDMs)
#pragma unroll
        for (int i = 0; i < 8; ++i) {
            int idx = tid + i * 256;
            int r = idx >> 4, c4 = idx & 15;
            float4 f = *(const float4*)&Wo[(size_t)(nt * 128 + r) * EMBED + kc + c4 * 4];
            int o = r * 64 + c4 * 4;
            sB[o + 0] = f2bf(f.x); sB[o + 1] = f2bf(f.y);
            sB[o + 2] = f2bf(f.z); sB[o + 3] = f2bf(f.w);
        }
#if HAVE_TDM
        if (wave == 0) {
            if (kc + 64 < EMBED) __builtin_amdgcn_s_wait_tensorcnt((short)1);
            else                 __builtin_amdgcn_s_wait_tensorcnt((short)0);
        }
#endif
        __syncthreads();

#pragma unroll
        for (int ks = 0; ks < 64; ks += 32) {
            v16bf af[2], bf[2];
#pragma unroll
            for (int i = 0; i < 2; ++i)
                af[i] = load_afrag(&sA[cur][0], 64, wm * 32 + i * 16 + hl, ks, g);
#pragma unroll
            for (int j = 0; j < 2; ++j)
                bf[j] = load_bfrag(sB, 64, wn * 32 + j * 16 + hl, ks, g);
#pragma unroll
            for (int i = 0; i < 2; ++i)
#pragma unroll
                for (int j = 0; j < 2; ++j)
                    acc[i][j] = wmma_bf16(af[i], bf[j], acc[i][j]);
        }
        __syncthreads();
    }

    const int c = lane & 15;
#pragma unroll
    for (int i = 0; i < 2; ++i) {
#pragma unroll
        for (int j = 0; j < 2; ++j) {
            int n = nt * 128 + wn * 32 + j * 16 + c;
            float bn = bo[n];
#pragma unroll
            for (int r = 0; r < 8; ++r) {
                int row = mt * 64 + wm * 32 + i * 16 + r + 8 * g;
                size_t o = (size_t)row * EMBED + n;
                tmp[o] = acc[i][j][r] + bn + X[o];
            }
        }
    }
}

// ---------------------------------------------------------------------------
// Kernel 4: LayerNorm over last dim (1024).  One 256-thread block per row.
// ---------------------------------------------------------------------------
__global__ __launch_bounds__(256) void ln_kernel(
    const float* __restrict__ tmp, const float* __restrict__ gamma,
    const float* __restrict__ beta, float* __restrict__ out)
{
    const int row = blockIdx.x;
    const int tid = threadIdx.x, lane = tid & 31, wave = tid >> 5;
    const float* x = tmp + (size_t)row * EMBED;

    float4 v = *(const float4*)&x[tid * 4];
    float s  = v.x + v.y + v.z + v.w;
    float s2 = v.x * v.x + v.y * v.y + v.z * v.z + v.w * v.w;
#pragma unroll
    for (int m = 1; m < 32; m <<= 1) {
        s  += __shfl_xor(s,  m, 32);
        s2 += __shfl_xor(s2, m, 32);
    }
    __shared__ float rs[8], rs2[8];
    if (lane == 0) { rs[wave] = s; rs2[wave] = s2; }
    __syncthreads();
    if (wave == 0) {
        float a  = (lane < 8) ? rs[lane]  : 0.0f;
        float a2 = (lane < 8) ? rs2[lane] : 0.0f;
#pragma unroll
        for (int m = 1; m < 8; m <<= 1) {
            a  += __shfl_xor(a,  m, 32);
            a2 += __shfl_xor(a2, m, 32);
        }
        if (lane == 0) { rs[0] = a; rs2[0] = a2; }
    }
    __syncthreads();
    const float mean = rs[0] * (1.0f / EMBED);
    const float var  = rs2[0] * (1.0f / EMBED) - mean * mean;
    const float rstd = rsqrtf(var + 1e-12f);

    const int c = tid * 4;
    float4 gm = *(const float4*)&gamma[c];
    float4 bt = *(const float4*)&beta[c];
    float4 o;
    o.x = (v.x - mean) * rstd * gm.x + bt.x;
    o.y = (v.y - mean) * rstd * gm.y + bt.y;
    o.z = (v.z - mean) * rstd * gm.z + bt.z;
    o.w = (v.w - mean) * rstd * gm.w + bt.w;
    *(float4*)&out[(size_t)row * EMBED + c] = o;
}

// ---------------------------------------------------------------------------
extern "C" void kernel_launch(void* const* d_in, const int* in_sizes, int n_in,
                              void* d_out, int out_size, void* d_ws, size_t ws_size,
                              hipStream_t stream)
{
    const float* X     = (const float*)d_in[0];
    const float* mask  = (const float*)d_in[1];
    const float* Wq    = (const float*)d_in[2];
    const float* bq    = (const float*)d_in[3];
    const float* Wk    = (const float*)d_in[4];
    const float* bk    = (const float*)d_in[5];
    const float* Wv    = (const float*)d_in[6];
    const float* bv    = (const float*)d_in[7];
    const float* Wo    = (const float*)d_in[8];
    const float* bo    = (const float*)d_in[9];
    const float* gamma = (const float*)d_in[10];
    const float* beta  = (const float*)d_in[11];
    float* out = (float*)d_out;

    // Workspace carving (48 MiB total).
    char* ws = (char*)d_ws;
    const size_t SZ_BF = (size_t)NTOK * EMBED * sizeof(unsigned short); // 8 MiB
    unsigned short* Qb  = (unsigned short*)(ws);
    unsigned short* Kb  = (unsigned short*)(ws + SZ_BF);
    unsigned short* Vt  = (unsigned short*)(ws + 2 * SZ_BF);
    unsigned short* ctx = (unsigned short*)(ws + 3 * SZ_BF);
    float*          tmp = (float*)(ws + 4 * SZ_BF);                     // 16 MiB

    qkv_proj_kernel<<<dim3(NTOK / 64, EMBED / 128, 3), 256, 0, stream>>>(
        X, Wq, bq, Wk, bk, Wv, bv, Qb, Kb, Vt);

    attn_kernel<<<dim3(SEQ / 64, BATCH * HEADS), 128, 0, stream>>>(
        Qb, Kb, Vt, mask, ctx);

    oproj_kernel<<<dim3(NTOK / 64, EMBED / 128), 256, 0, stream>>>(
        ctx, Wo, bo, X, tmp);

    ln_kernel<<<NTOK, 256, 0, stream>>>(tmp, gamma, beta, out);
}
